// ParallelTransformer_79156247265926
// MI455X (gfx1250) — matrix operand order repaired and checked by reading, not compiled, not run on hardware
//
#include <hip/hip_runtime.h>
#include <hip/hip_bf16.h>

typedef __attribute__((ext_vector_type(16))) _Float16 v16h;
typedef __attribute__((ext_vector_type(8)))  float    v8f;
typedef __attribute__((ext_vector_type(4)))  float    f32x4;
typedef __attribute__((ext_vector_type(4)))  unsigned u32x4;
typedef __attribute__((ext_vector_type(8)))  int      i32x8;
typedef __attribute__((ext_vector_type(4)))  int      i32x4;

// Address-space-qualified vector pointers for the async-LDS builtin
// (diagnostic showed param type: v4i in addrspace(1)/__device__).
typedef __attribute__((address_space(1))) i32x4 as1_i32x4;
typedef __attribute__((address_space(3))) i32x4 as3_i32x4;

// ---- CDNA5 feature probes (host pass sees 0; device pass enables) ----------
#if __has_builtin(__builtin_amdgcn_global_load_async_to_lds_b128)
#define HAVE_ASYNC_LDS 1
#else
#define HAVE_ASYNC_LDS 0
#endif

#if __has_builtin(__builtin_amdgcn_tensor_load_to_lds)
#define HAVE_TDM 1
#else
#define HAVE_TDM 0
#endif

// ---------------------------------------------------------------------------
// Wait helpers (counted waits need immediates -> template constants)
// ---------------------------------------------------------------------------
template <int N>
__device__ __forceinline__ void wait_async_c() {
#if HAVE_ASYNC_LDS
#if __has_builtin(__builtin_amdgcn_s_wait_asynccnt)
  __builtin_amdgcn_s_wait_asynccnt(N);
#else
  asm volatile("s_wait_asynccnt %0" ::"i"(N) : "memory");
#endif
#endif
}

__device__ __forceinline__ void wait_tensor0() {
#if HAVE_TDM
#if __has_builtin(__builtin_amdgcn_s_wait_tensorcnt)
  __builtin_amdgcn_s_wait_tensorcnt(0);
#else
  asm volatile("s_wait_tensorcnt 0" ::: "memory");
#endif
#endif
}

// ---------------------------------------------------------------------------
// Async global->LDS 16-byte copy (ASYNCcnt); sync fallback if builtin absent.
// ---------------------------------------------------------------------------
__device__ __forceinline__ void async_cp_b128(const _Float16* g, _Float16* l) {
#if HAVE_ASYNC_LDS
  __builtin_amdgcn_global_load_async_to_lds_b128(
      (as1_i32x4*)(uintptr_t)g, (as3_i32x4*)(uintptr_t)l, 0, 0);
#else
  *(f32x4*)l = *(const f32x4*)g;
#endif
}

// ---------------------------------------------------------------------------
// TDM: 2D tile load, rows x 64 halves, into LDS with an 8-half pad per row
// (pad_interval = 32 DWORDs -> code 4, pad_amount = 4 DWORDs -> code 3),
// matching an LDS row stride of 72 halves. data_size = 2 bytes (code 1).
// D# bit-packing per CDNA5 ISA.
// ---------------------------------------------------------------------------
#if HAVE_TDM
__device__ __forceinline__ void tdm_load_rows64_f16(const _Float16* g,
                                                    unsigned lds_byte_addr,
                                                    int rows, int stride_elems) {
  unsigned long long ga = (unsigned long long)(uintptr_t)g;
  u32x4 g0;
  g0[0] = 1u;                                            // count=1, user mode
  g0[1] = lds_byte_addr;                                 // lds_addr
  g0[2] = (unsigned)ga;                                  // global_addr lo
  g0[3] = (unsigned)((ga >> 32) & 0x01FFFFFFu) | (2u << 30);  // addr hi | type=2
  i32x8 g1;
  g1[0] = (int)((1u << 16) | (1u << 20) | (4u << 22) | (3u << 25));
  //            data_size=2B  pad_enable  interval=32DW  amount=4DW
  unsigned d0 = (unsigned)stride_elems;   // tensor_dim0 (elements per row)
  unsigned d1 = (unsigned)rows;           // tensor_dim1
  g1[1] = (int)((d0 & 0xFFFFu) << 16);                       // dim0[15:0]
  g1[2] = (int)(((d0 >> 16) & 0xFFFFu) | ((d1 & 0xFFFFu) << 16));
  g1[3] = (int)(((d1 >> 16) & 0xFFFFu) | (64u << 16));       // tile_dim0 = 64
  g1[4] = (int)(d1 & 0xFFFFu);                               // tile_dim1 = rows
  g1[5] = stride_elems;                                      // dim0_stride lo32
  g1[6] = 0;
  g1[7] = 0;
  i32x4 z4 = {};
#if __clang_major__ >= 23
  i32x8 z8 = {};
  __builtin_amdgcn_tensor_load_to_lds(g0, g1, z4, z4, z8, 0);
#else
  __builtin_amdgcn_tensor_load_to_lds(g0, g1, z4, z4, 0);
#endif
}
#endif

// ---------------------------------------------------------------------------
// WMMA fragment helpers (CDNA5 wave32, 16x16x32 f16 -> f32)
// ---------------------------------------------------------------------------
__device__ __forceinline__ v16h frag_a(const _Float16* p, int ld, int ln, int hi) {
  const _Float16* r = p + ln * ld + hi * 16;
  v16h f;
#pragma unroll
  for (int i = 0; i < 16; ++i) f[i] = r[i];
  return f;
}

__device__ __forceinline__ v16h frag_b(const _Float16* p, int ld, int ln, int hi) {
  const _Float16* r = p + ln * ld + hi * 16;
  v16h f;
#pragma unroll
  for (int i = 0; i < 16; ++i) f[i] = r[i];
  return f;
}

__device__ __forceinline__ v8f wmma_f16(v16h a, v16h b, v8f c) {
  return __builtin_amdgcn_wmma_f32_16x16x32_f16(false, a, false, b, (short)0, c,
                                                false, false);
}

__device__ __forceinline__ float gelu_tanh(float x) {
  return 0.5f * x * (1.0f + tanhf(0.7978845608028654f * (x + 0.044715f * x * x * x)));
}

// ---------------------------------------------------------------------------
// RMSNorm: one block per row, fp32 in -> f16 out
// ---------------------------------------------------------------------------
__global__ __launch_bounds__(256)
void rmsnorm_f16_kernel(const float* __restrict__ x, const float* __restrict__ w,
                        _Float16* __restrict__ out, int cols) {
  const int row = blockIdx.x;
  const float* xr = x + (size_t)row * cols;
  float s = 0.f;
  for (int c = threadIdx.x; c < cols; c += blockDim.x) {
    float v = xr[c];
    s += v * v;
  }
#pragma unroll
  for (int off = 16; off > 0; off >>= 1) s += __shfl_xor(s, off, 32);
  __shared__ float ps[8];
  if ((threadIdx.x & 31) == 0) ps[threadIdx.x >> 5] = s;
  __syncthreads();
  float tot = 0.f;
#pragma unroll
  for (int i = 0; i < 8; ++i) tot += ps[i];
  const float scale = rsqrtf(tot / (float)cols + 1e-6f);
  for (int c = threadIdx.x; c < cols; c += blockDim.x)
    out[(size_t)row * cols + c] = (_Float16)(w[c] * xr[c] * scale);
}

// ---------------------------------------------------------------------------
// Weight convert + transpose: w[K][N] fp32 -> wt[N][K] f16
// ---------------------------------------------------------------------------
__global__ __launch_bounds__(256)
void transpose_f16_kernel(const float* __restrict__ w, _Float16* __restrict__ wt,
                          int K, int N) {
  __shared__ _Float16 tile[32][33];
  const int n0 = blockIdx.x * 32, k0 = blockIdx.y * 32;
  const int tx = threadIdx.x & 31, ty = threadIdx.x >> 5;
  for (int i = ty; i < 32; i += 8)
    tile[i][tx] = (_Float16)w[(size_t)(k0 + i) * N + n0 + tx];
  __syncthreads();
  for (int i = ty; i < 32; i += 8)
    wt[(size_t)(n0 + i) * K + k0 + tx] = tile[tx][i];
}

// ---------------------------------------------------------------------------
// WMMA GEMM: C[M,N] = A[M,K](f16) x Bt[N,K](f16). Block tile 64x128, BK=64,
// double-buffered LDS staged with async global->LDS (6 b128/wave/stage;
// pipeline: issue stage s+1, s_wait_asynccnt 6 -> stage s resident).
// Epilogue: optional bias[n], tanh-GELU, residual[m*N+n]; f32 and/or f16 out.
// ---------------------------------------------------------------------------
__global__ __launch_bounds__(256)
void gemm_wmma_kernel(const _Float16* __restrict__ A, const _Float16* __restrict__ Bt,
                      float* __restrict__ outF, _Float16* __restrict__ outH,
                      const float* __restrict__ bias, const float* __restrict__ residual,
                      int M, int N, int K, int apply_gelu) {
  constexpr int LDT = 72;
  __shared__ _Float16 As[2][64][LDT];    // 18 KB
  __shared__ _Float16 Bs[2][128][LDT];   // 36 KB
  const int tid = threadIdx.x;
  const int lane = tid & 31, wv = tid >> 5;
  const int ln = lane & 15, hi = lane >> 4;
  const int wm = wv >> 2, wn = wv & 3;
  const int m0 = blockIdx.y * 64, n0 = blockIdx.x * 128;

  v8f zero8 = {};
  v8f c00 = zero8, c01 = zero8, c10 = zero8, c11 = zero8;

  const int arow = tid >> 2, acol = (tid & 3) * 16;  // 64 rows x 64 halves
  const int brow = tid >> 1, bcol = (tid & 1) * 32;  // 128 rows x 64 halves

  const _Float16* gA = A + (size_t)(m0 + arow) * K + acol;
  const _Float16* gB = Bt + (size_t)(n0 + brow) * K + bcol;

  auto stage = [&](int s, int k0) {
    async_cp_b128(gA + k0,      &As[s][arow][acol]);
    async_cp_b128(gA + k0 + 8,  &As[s][arow][acol + 8]);
    async_cp_b128(gB + k0,      &Bs[s][brow][bcol]);
    async_cp_b128(gB + k0 + 8,  &Bs[s][brow][bcol + 8]);
    async_cp_b128(gB + k0 + 16, &Bs[s][brow][bcol + 16]);
    async_cp_b128(gB + k0 + 24, &Bs[s][brow][bcol + 24]);
  };

  stage(0, 0);
  int s = 0;
  for (int k0 = 0; k0 < K; k0 += 64, s ^= 1) {
    if (k0 + 64 < K) {
      stage(s ^ 1, k0 + 64);   // prefetch next stage
      wait_async_c<6>();       // current stage resident; next stays in flight
    } else {
      wait_async_c<0>();
    }
    __syncthreads();
#pragma unroll
    for (int ks = 0; ks < 2; ++ks) {
      v16h a0 = frag_a(&As[s][wm * 32][ks * 32], LDT, ln, hi);
      v16h a1 = frag_a(&As[s][wm * 32 + 16][ks * 32], LDT, ln, hi);
      v16h b0 = frag_b(&Bs[s][wn * 32][ks * 32], LDT, ln, hi);
      v16h b1 = frag_b(&Bs[s][wn * 32 + 16][ks * 32], LDT, ln, hi);
      c00 = wmma_f16(a0, b0, c00);
      c01 = wmma_f16(a0, b1, c01);
      c10 = wmma_f16(a1, b0, c10);
      c11 = wmma_f16(a1, b1, c11);
    }
    __syncthreads();
  }

  auto store_frag = [&](v8f c, int mi, int ni) {
    const int gm = m0 + wm * 32 + mi * 16 + hi * 8;
    const int gn = n0 + wn * 32 + ni * 16 + ln;
#pragma unroll
    for (int r = 0; r < 8; ++r) {
      const size_t idx = (size_t)(gm + r) * N + gn;
      float v = c[r];
      if (bias) v += bias[gn];
      if (apply_gelu) v = gelu_tanh(v);
      if (residual) v += residual[idx];
      if (outF) outF[idx] = v;
      if (outH) outH[idx] = (_Float16)v;
    }
  };
  store_frag(c00, 0, 0);
  store_frag(c01, 0, 1);
  store_frag(c10, 1, 0);
  store_frag(c11, 1, 1);
}

// ---------------------------------------------------------------------------
// Flash attention (f16 QKV in, f32 online softmax, f16 ctx out).
// Q tile (128x64) and each K tile (64x64) are staged by the Tensor Data Mover
// (pad_enable gives the 72-half LDS row stride); V is loaded transposed by the
// block. Wave-local online softmax; P goes through wave-private LDS to become
// an A-fragment for the P·V WMMA.
// ---------------------------------------------------------------------------
__global__ __launch_bounds__(256)
void flash_attn_kernel(const _Float16* __restrict__ Q, const _Float16* __restrict__ Kg,
                       const _Float16* __restrict__ V, const float* __restrict__ mask,
                       _Float16* __restrict__ ctx, int S, int Hn, int Dm) {
  constexpr int QT = 128, KT = 64, DH = 64, LD = 72;
  __shared__ _Float16 Qs[QT][LD];        // 18 KB
  __shared__ _Float16 Ks[KT][LD];        //  9 KB
  __shared__ _Float16 Vt[DH][LD];        //  9 KB (transposed [dh][key])
  __shared__ _Float16 Ps[8][16][LD];     // 18 KB (per-wave P staging)

  const int b = blockIdx.y / Hn, h = blockIdx.y % Hn;
  const int q0 = blockIdx.x * QT;
  const int tid = threadIdx.x, lane = tid & 31, wv = tid >> 5;
  const int ln = lane & 15, hi = lane >> 4;

  // Stage Q tile
#if HAVE_TDM
  if (wv == 0) {
    tdm_load_rows64_f16(&Q[((size_t)(b * S + q0)) * Dm + h * DH],
                        (unsigned)(uintptr_t)&Qs[0][0], QT, Dm);
    wait_tensor0();
  }
#else
  for (int i = tid; i < QT * DH / 8; i += 256) {
    const int r = i >> 3, c = (i & 7) * 8;
    *(f32x4*)&Qs[r][c] =
        *(const f32x4*)&Q[((size_t)(b * S + q0 + r)) * Dm + h * DH + c];
  }
#endif
  __syncthreads();

  v8f zero8 = {};
  v8f oAcc[4] = {zero8, zero8, zero8, zero8};
  float rm[8], rl[8];
#pragma unroll
  for (int r = 0; r < 8; ++r) { rm[r] = -1e30f; rl[r] = 0.f; }

  for (int kt = 0; kt < S; kt += KT) {
    // Stage K tile (TDM) + transposed V tile (manual)
#if HAVE_TDM
    if (wv == 0)
      tdm_load_rows64_f16(&Kg[((size_t)(b * S + kt)) * Dm + h * DH],
                          (unsigned)(uintptr_t)&Ks[0][0], KT, Dm);
#endif
    for (int i = tid; i < KT * DH / 8; i += 256) {
      const int r = i >> 3, c = (i & 7) * 8;
#if !HAVE_TDM
      *(f32x4*)&Ks[r][c] =
          *(const f32x4*)&Kg[((size_t)(b * S + kt + r)) * Dm + h * DH + c];
#endif
      f32x4 tmp = *(const f32x4*)&V[((size_t)(b * S + kt + r)) * Dm + h * DH + c];
      const _Float16* tv = (const _Float16*)&tmp;
#pragma unroll
      for (int j = 0; j < 8; ++j) Vt[c + j][r] = tv[j];
    }
#if HAVE_TDM
    if (wv == 0) wait_tensor0();
#endif
    __syncthreads();

    // --- scores: this wave's 16 q-rows x 64 keys (K-dim = DH, 2 steps)
    v8f sAcc[4] = {zero8, zero8, zero8, zero8};
#pragma unroll
    for (int ks = 0; ks < 2; ++ks) {
      v16h a = frag_a(&Qs[wv * 16][ks * 32], LD, ln, hi);
#pragma unroll
      for (int nf = 0; nf < 4; ++nf) {
        v16h bf = frag_b(&Ks[nf * 16][ks * 32], LD, ln, hi);
        sAcc[nf] = wmma_f16(a, bf, sAcc[nf]);
      }
    }

    float sc[4][8];
#pragma unroll
    for (int nf = 0; nf < 4; ++nf) {
      const float mb = (mask[(size_t)b * S + kt + nf * 16 + ln] - 1.f) * 1e9f;
#pragma unroll
      for (int r = 0; r < 8; ++r) sc[nf][r] = sAcc[nf][r] * 0.125f + mb;
    }

    float alpha[8];
#pragma unroll
    for (int r = 0; r < 8; ++r) {
      float mx = fmaxf(fmaxf(sc[0][r], sc[1][r]), fmaxf(sc[2][r], sc[3][r]));
#pragma unroll
      for (int off = 8; off > 0; off >>= 1) mx = fmaxf(mx, __shfl_xor(mx, off, 32));
      const float mn = fmaxf(rm[r], mx);
      alpha[r] = __expf(rm[r] - mn);
      rm[r] = mn;
    }

    float psum[8];
#pragma unroll
    for (int r = 0; r < 8; ++r) psum[r] = 0.f;
#pragma unroll
    for (int nf = 0; nf < 4; ++nf)
#pragma unroll
      for (int r = 0; r < 8; ++r) {
        const float p = __expf(sc[nf][r] - rm[r]);
        psum[r] += p;
        Ps[wv][hi * 8 + r][nf * 16 + ln] = (_Float16)p;  // C-layout -> A-layout
      }
#pragma unroll
    for (int r = 0; r < 8; ++r) {
      float t = psum[r];
#pragma unroll
      for (int off = 8; off > 0; off >>= 1) t += __shfl_xor(t, off, 32);
      rl[r] = rl[r] * alpha[r] + t;
    }
#pragma unroll
    for (int nf = 0; nf < 4; ++nf)
#pragma unroll
      for (int r = 0; r < 8; ++r) oAcc[nf][r] *= alpha[r];

    // --- P(16x64) x V(64x64)
#pragma unroll
    for (int ks = 0; ks < 2; ++ks) {
      v16h a = frag_a(&Ps[wv][0][ks * 32], LD, ln, hi);
#pragma unroll
      for (int nf = 0; nf < 4; ++nf) {
        v16h bf = frag_b(&Vt[nf * 16][ks * 32], LD, ln, hi);
        oAcc[nf] = wmma_f16(a, bf, oAcc[nf]);
      }
    }
    __syncthreads();
  }

#pragma unroll
  for (int nf = 0; nf < 4; ++nf)
#pragma unroll
    for (int r = 0; r < 8; ++r) {
      const int gq = q0 + wv * 16 + hi * 8 + r;
      const int gd = nf * 16 + ln;
      const float denom = fmaxf(rl[r], 1e-20f);
      ctx[((size_t)(b * S + gq)) * Dm + h * DH + gd] = (_Float16)(oAcc[nf][r] / denom);
    }
}

// ---------------------------------------------------------------------------
// Host-side orchestration
// ---------------------------------------------------------------------------
extern "C" void kernel_launch(void* const* d_in, const int* in_sizes, int n_in,
                              void* d_out, int out_size, void* d_ws, size_t ws_size,
                              hipStream_t stream) {
  (void)in_sizes; (void)n_in; (void)out_size; (void)ws_size;
  const float* X    = (const float*)d_in[0];
  const float* mask = (const float*)d_in[1];
  const float* n1w  = (const float*)d_in[2];
  const float* n2w  = (const float*)d_in[3];
  const float* Wq   = (const float*)d_in[4];
  const float* Wk   = (const float*)d_in[5];
  const float* Wv   = (const float*)d_in[6];
  const float* Wo   = (const float*)d_in[7];
  const float* ff1w = (const float*)d_in[8];
  const float* ff1b = (const float*)d_in[9];
  const float* ff2w = (const float*)d_in[10];
  const float* ff2b = (const float*)d_in[11];
  float* out = (float*)d_out;

  const int Bn = 2, S = 2048, D = 1024, Hn = 16, FF = 4096;
  const int M = Bn * S;  // 4096 rows

  char* base = (char*)d_ws;
  size_t off = 0;
  auto alloc = [&](size_t bytes) -> char* {
    char* p = base + off;
    off = (off + bytes + 255) & ~(size_t)255;
    return p;
  };
  _Float16* Xn   = (_Float16*)alloc((size_t)M * D * 2);
  _Float16* WqT  = (_Float16*)alloc((size_t)D * D * 2);
  _Float16* WkT  = (_Float16*)alloc((size_t)D * D * 2);
  _Float16* WvT  = (_Float16*)alloc((size_t)D * D * 2);
  _Float16* WoT  = (_Float16*)alloc((size_t)D * D * 2);
  _Float16* ff1T = (_Float16*)alloc((size_t)FF * D * 2);
  _Float16* ff2T = (_Float16*)alloc((size_t)D * FF * 2);
  _Float16* Qh   = (_Float16*)alloc((size_t)M * D * 2);
  _Float16* Kh   = (_Float16*)alloc((size_t)M * D * 2);
  _Float16* Vh   = (_Float16*)alloc((size_t)M * D * 2);
  _Float16* Ctx  = (_Float16*)alloc((size_t)M * D * 2);
  float*    X1   = (float*)   alloc((size_t)M * D * 4);
  _Float16* Xn2  = (_Float16*)alloc((size_t)M * D * 2);
  _Float16* Hf   = (_Float16*)alloc((size_t)M * FF * 2);

  const dim3 blk(256);

  transpose_f16_kernel<<<dim3(D / 32, D / 32), blk, 0, stream>>>(Wq, WqT, D, D);
  transpose_f16_kernel<<<dim3(D / 32, D / 32), blk, 0, stream>>>(Wk, WkT, D, D);
  transpose_f16_kernel<<<dim3(D / 32, D / 32), blk, 0, stream>>>(Wv, WvT, D, D);
  transpose_f16_kernel<<<dim3(D / 32, D / 32), blk, 0, stream>>>(Wo, WoT, D, D);
  transpose_f16_kernel<<<dim3(FF / 32, D / 32), blk, 0, stream>>>(ff1w, ff1T, D, FF);
  transpose_f16_kernel<<<dim3(D / 32, FF / 32), blk, 0, stream>>>(ff2w, ff2T, FF, D);

  rmsnorm_f16_kernel<<<dim3(M), blk, 0, stream>>>(X, n1w, Xn, D);

  gemm_wmma_kernel<<<dim3(D / 128, M / 64), blk, 0, stream>>>(
      Xn, WqT, nullptr, Qh, nullptr, nullptr, M, D, D, 0);
  gemm_wmma_kernel<<<dim3(D / 128, M / 64), blk, 0, stream>>>(
      Xn, WkT, nullptr, Kh, nullptr, nullptr, M, D, D, 0);
  gemm_wmma_kernel<<<dim3(D / 128, M / 64), blk, 0, stream>>>(
      Xn, WvT, nullptr, Vh, nullptr, nullptr, M, D, D, 0);

  flash_attn_kernel<<<dim3(S / 128, Bn * Hn), blk, 0, stream>>>(
      Qh, Kh, Vh, mask, Ctx, S, Hn, D);

  gemm_wmma_kernel<<<dim3(D / 128, M / 64), blk, 0, stream>>>(
      Ctx, WoT, X1, nullptr, nullptr, X, M, D, D, 0);

  rmsnorm_f16_kernel<<<dim3(M), blk, 0, stream>>>(X1, n2w, Xn2, D);

  gemm_wmma_kernel<<<dim3(FF / 128, M / 64), blk, 0, stream>>>(
      Xn2, ff1T, nullptr, Hf, ff1b, nullptr, M, FF, D, 1);

  gemm_wmma_kernel<<<dim3(D / 128, M / 64), blk, 0, stream>>>(
      Hf, ff2T, out, nullptr, ff2b, X1, M, D, FF, 0);
}
